// TransformerBlock_71777493451131
// MI455X (gfx1250) — matrix operand order repaired and checked by reading, not compiled, not run on hardware
//
#include <hip/hip_runtime.h>
#include <hip/hip_bf16.h>

typedef unsigned short u16;
typedef unsigned int   u32;
typedef __attribute__((ext_vector_type(16))) __bf16 v16bf;
typedef __attribute__((ext_vector_type(8)))  float  v8f;

union FragBF { v16bf v; uint4 u[2]; };

__device__ __forceinline__ u16 f2bf(float f) {
    union { float f; unsigned u; } c; c.f = f;
    unsigned r = c.u + 0x7FFFu + ((c.u >> 16) & 1u);
    return (u16)(r >> 16);
}
__device__ __forceinline__ u32 pack2bf(float lo, float hi) {
    return (u32)f2bf(lo) | ((u32)f2bf(hi) << 16);
}

__device__ __forceinline__ v8f wmma_bf16(const FragBF& a, const FragBF& b, v8f c) {
    return __builtin_amdgcn_wmma_f32_16x16x32_bf16(
        /*neg_a=*/false, a.v, /*neg_b=*/false, b.v,
        /*c_mod=*/(short)0, c, /*reuse_a=*/false, /*reuse_b=*/false);
}

// ---------------------------------------------------------------------------
// LayerNorm: one 256-thread block per row of 1024 fp32, bf16 output.
// ---------------------------------------------------------------------------
__global__ __launch_bounds__(256) void layernorm_bf16(
    const float* __restrict__ x, const float* __restrict__ g,
    const float* __restrict__ b, u16* __restrict__ out) {
    constexpr int C = 1024;
    __shared__ float red[256];
    const int row = blockIdx.x, t = threadIdx.x;
    const float* xr = x + (size_t)row * C;
    float v[4], s = 0.f;
#pragma unroll
    for (int i = 0; i < 4; i++) { v[i] = xr[t + 256 * i]; s += v[i]; }
    red[t] = s; __syncthreads();
    for (int o = 128; o > 0; o >>= 1) { if (t < o) red[t] += red[t + o]; __syncthreads(); }
    const float mu = red[0] * (1.f / C);
    __syncthreads();
    float s2 = 0.f;
#pragma unroll
    for (int i = 0; i < 4; i++) { float d = v[i] - mu; s2 += d * d; }
    red[t] = s2; __syncthreads();
    for (int o = 128; o > 0; o >>= 1) { if (t < o) red[t] += red[t + o]; __syncthreads(); }
    const float rstd = rsqrtf(red[0] * (1.f / C) + 1e-5f);
    u16* orow = out + (size_t)row * C;
#pragma unroll
    for (int i = 0; i < 4; i++) {
        int c = t + 256 * i;
        orow[c] = f2bf((v[i] - mu) * rstd * g[c] + b[c]);
    }
}

// ---------------------------------------------------------------------------
// GEMM: C = epilogue(A[M,K](bf16) @ W[K,N](fp32) + bias)
// 256 threads = 8 waves, 128x128 block tile, K-step 32, wave tile 32x64.
// Double-buffered LDS: next tile's global loads issue before the WMMA block,
// one barrier per K-step. W is converted fp32->bf16 and stored transposed
// [n][k] with (k,k+1) pairs packed into dword stores.
// MODE 0: bf16 out (bias)   MODE 1: f32 out (bias + residual)
// MODE 2: bf16 out (bias + exact GELU)
// ---------------------------------------------------------------------------
template <int MODE>
__global__ __launch_bounds__(256) void gemm_bf16(
    const u16* __restrict__ A, const float* __restrict__ Wf,
    const float* __restrict__ bias, const float* __restrict__ res,
    void* __restrict__ out, int M, int N, int K) {
    constexpr int TM = 128, TN = 128, TK = 32;
    __shared__ u16 Al[2][TM * TK];   // row-major [m][k]
    __shared__ u16 Bl[2][TN * TK];   // transposed [n][k]

    const int bn = blockIdx.x * TN, bm = blockIdx.y * TM;
    const int t = threadIdx.x, wid = t >> 5, lane = t & 31;
    const int wm = (wid >> 1) * 32, wn = (wid & 1) * 64;
    const int half = lane >> 4, l16 = lane & 15;

    // A staging: chunk = 8 contiguous bf16 of one row. 512 chunks, 2/thread.
    int aRow[2], aCol[2];
    // W staging: chunk = 2 k-rows x 4 n-cols of fp32. 512 chunks, 2/thread.
    int wKr[2], wNc[2];
#pragma unroll
    for (int i = 0; i < 2; i++) {
        int chunk = t + 256 * i;
        aRow[i] = chunk >> 2;           // 0..127
        aCol[i] = (chunk & 3) * 8;      // 0,8,16,24
        wKr[i] = (chunk >> 5) * 2;      // 0..30 (even)
        wNc[i] = (chunk & 31) * 4;      // 0..124
    }

    uint4  aReg[2];
    float4 wReg[2][2];

    auto loadGlobal = [&](int k0) {
#pragma unroll
        for (int i = 0; i < 2; i++) {
            aReg[i] = *(const uint4*)&A[(size_t)(bm + aRow[i]) * K + k0 + aCol[i]];
            wReg[i][0] = *(const float4*)&Wf[(size_t)(k0 + wKr[i]) * N + bn + wNc[i]];
            wReg[i][1] = *(const float4*)&Wf[(size_t)(k0 + wKr[i] + 1) * N + bn + wNc[i]];
        }
    };
    auto storeLDS = [&](int buf) {
#pragma unroll
        for (int i = 0; i < 2; i++) {
            *(uint4*)&Al[buf][aRow[i] * TK + aCol[i]] = aReg[i];
            const float* w0 = (const float*)&wReg[i][0];
            const float* w1 = (const float*)&wReg[i][1];
#pragma unroll
            for (int j = 0; j < 4; j++)
                *(u32*)&Bl[buf][(wNc[i] + j) * TK + wKr[i]] = pack2bf(w0[j], w1[j]);
        }
    };

    v8f acc[2][4] = {};

    const int nk = K / TK;
    loadGlobal(0);
    storeLDS(0);
    __syncthreads();

    for (int kt = 0; kt < nk; kt++) {
        const int buf = kt & 1;
        if (kt + 1 < nk) loadGlobal((kt + 1) * TK);  // overlap with WMMA below

        FragBF bfr[4];
#pragma unroll
        for (int nt = 0; nt < 4; nt++) {
            int col = wn + nt * 16 + l16, ks = half * 8;
            bfr[nt].u[0] = *(const uint4*)&Bl[buf][col * TK + ks];
            bfr[nt].u[1] = *(const uint4*)&Bl[buf][col * TK + ks + 16];
        }
#pragma unroll
        for (int mt = 0; mt < 2; mt++) {
            FragBF a;
            int row = wm + mt * 16 + l16, kb = half * 8;
            a.u[0] = *(const uint4*)&Al[buf][row * TK + kb];
            a.u[1] = *(const uint4*)&Al[buf][row * TK + kb + 16];
#pragma unroll
            for (int nt = 0; nt < 4; nt++)
                acc[mt][nt] = wmma_bf16(a, bfr[nt], acc[mt][nt]);
        }
        if (kt + 1 < nk) storeLDS(buf ^ 1);
        __syncthreads();
    }

#pragma unroll
    for (int mt = 0; mt < 2; mt++)
#pragma unroll
        for (int nt = 0; nt < 4; nt++) {
            int col = bn + wn + nt * 16 + l16;
            float bz = bias[col];
#pragma unroll
            for (int v = 0; v < 8; v++) {
                int row = bm + wm + mt * 16 + v + half * 8;
                size_t idx = (size_t)row * N + col;
                float val = acc[mt][nt][v] + bz;
                if (MODE == 0) {
                    ((u16*)out)[idx] = f2bf(val);
                } else if (MODE == 1) {
                    ((float*)out)[idx] = val + res[idx];
                } else {
                    float ge = 0.5f * val * (1.f + erff(val * 0.70710678f));
                    ((u16*)out)[idx] = f2bf(ge);
                }
            }
        }
}

// ---------------------------------------------------------------------------
// Flash attention: grid (T/64, H, B), 128 threads = 4 waves, each wave owns
// 16 query rows. Dh = 64, causal, scale 1/8. Q/K/V bf16 [B*T, 1024].
// ---------------------------------------------------------------------------
__global__ __launch_bounds__(128) void flash_attn(
    const u16* __restrict__ Q, const u16* __restrict__ K,
    const u16* __restrict__ V, u16* __restrict__ O, int T) {
    constexpr int D = 64, C = 1024;
    __shared__ u16 Ql[64 * D];      // [q][d]
    __shared__ u16 Kl[64 * D];      // [key][d]   (contraction-contiguous for QK^T)
    __shared__ u16 Vt[D * 64];      // [d][key]   (contraction-contiguous for P@V)
    __shared__ u16 Pl[4][16 * 64];  // per-wave P tile [qrow][key]

    const int qt = blockIdx.x, h = blockIdx.y, b = blockIdx.z;
    const int t = threadIdx.x, wid = t >> 5, lane = t & 31;
    const int half = lane >> 4, l16 = lane & 15;
    const size_t rowbase = (size_t)b * T;
    const int colh = h * D, q0 = qt * 64;

    // stage Q tile once
#pragma unroll
    for (int i = 0; i < 4; i++) {
        int chunk = t + 128 * i;
        int r = chunk >> 3, c = (chunk & 7) * 8;
        *(uint4*)&Ql[r * D + c] =
            *(const uint4*)&Q[(rowbase + q0 + r) * C + colh + c];
    }

    v8f o[4] = {};
    float m[8], l[8];
#pragma unroll
    for (int v = 0; v < 8; v++) { m[v] = -1e30f; l[v] = 0.f; }

    for (int kt = 0; kt <= qt; kt++) {
        const int k0 = kt * 64;
        __syncthreads();  // prior compute done (also fences initial Q stage)
        // K tile: straight copy, 512 chunks of 8 bf16
#pragma unroll
        for (int i = 0; i < 4; i++) {
            int chunk = t + 128 * i;
            int r = chunk >> 3, c = (chunk & 7) * 8;
            *(uint4*)&Kl[r * D + c] =
                *(const uint4*)&K[(rowbase + k0 + r) * C + colh + c];
        }
        // V tile: transpose, pack key-pairs into dword stores. 256 chunks of
        // 2 keys x 8 d, 2/thread.
#pragma unroll
        for (int i = 0; i < 2; i++) {
            int chunk = t + 128 * i;
            int r2 = (chunk >> 3) * 2, c = (chunk & 7) * 8;
            uint4 d0 = *(const uint4*)&V[(rowbase + k0 + r2) * C + colh + c];
            uint4 d1 = *(const uint4*)&V[(rowbase + k0 + r2 + 1) * C + colh + c];
            const u16* p0 = (const u16*)&d0;
            const u16* p1 = (const u16*)&d1;
#pragma unroll
            for (int j = 0; j < 8; j++)
                *(u32*)&Vt[(c + j) * 64 + r2] = (u32)p0[j] | ((u32)p1[j] << 16);
        }
        __syncthreads();

        // S = Q @ K^T  (16 q-rows x 64 keys per wave)
        v8f s[4] = {};
#pragma unroll
        for (int ks = 0; ks < 2; ks++) {
            FragBF a;
            int row = wid * 16 + l16, kb = ks * 32 + half * 8;
            a.u[0] = *(const uint4*)&Ql[row * D + kb];
            a.u[1] = *(const uint4*)&Ql[row * D + kb + 16];
#pragma unroll
            for (int nt = 0; nt < 4; nt++) {
                FragBF bb;
                int col = nt * 16 + l16, kss = ks * 32 + half * 8;
                bb.u[0] = *(const uint4*)&Kl[col * D + kss];
                bb.u[1] = *(const uint4*)&Kl[col * D + kss + 16];
                s[nt] = wmma_bf16(a, bb, s[nt]);
            }
        }

        // scale + causal mask + online softmax
        const bool diag = (kt == qt);
        float alpha[8];
#pragma unroll
        for (int v = 0; v < 8; v++) {
            int qrow = q0 + wid * 16 + v + half * 8;
            float mx = -1e30f;
#pragma unroll
            for (int nt = 0; nt < 4; nt++) {
                int kcol = k0 + nt * 16 + l16;
                float val = s[nt][v] * 0.125f;
                if (diag && kcol > qrow) val = -1e30f;
                s[nt][v] = val;
                mx = fmaxf(mx, val);
            }
#pragma unroll
            for (int d = 8; d >= 1; d >>= 1) mx = fmaxf(mx, __shfl_xor(mx, d, 32));
            float mt2 = fmaxf(m[v], mx);
            alpha[v] = __expf(m[v] - mt2);
            m[v] = mt2;
            float rs = 0.f;
#pragma unroll
            for (int nt = 0; nt < 4; nt++) {
                float p = __expf(s[nt][v] - mt2);
                s[nt][v] = p;
                rs += p;
            }
#pragma unroll
            for (int d = 8; d >= 1; d >>= 1) rs += __shfl_xor(rs, d, 32);
            l[v] = l[v] * alpha[v] + rs;
        }
#pragma unroll
        for (int nt = 0; nt < 4; nt++)
#pragma unroll
            for (int v = 0; v < 8; v++) o[nt][v] *= alpha[v];

        // P -> per-wave LDS (bf16) in A-ready row-major layout
#pragma unroll
        for (int v = 0; v < 8; v++) {
            int prow = v + 8 * half;
#pragma unroll
            for (int nt = 0; nt < 4; nt++)
                Pl[wid][prow * 64 + nt * 16 + l16] = f2bf(s[nt][v]);
        }

        // O += P @ V  (contraction over 64 keys)
#pragma unroll
        for (int ks = 0; ks < 2; ks++) {
            FragBF a;
            int kb = ks * 32 + half * 8;
            a.u[0] = *(const uint4*)&Pl[wid][l16 * 64 + kb];
            a.u[1] = *(const uint4*)&Pl[wid][l16 * 64 + kb + 16];
#pragma unroll
            for (int nt = 0; nt < 4; nt++) {
                FragBF bb;
                int col = nt * 16 + l16, kss = ks * 32 + half * 8;
                bb.u[0] = *(const uint4*)&Vt[col * 64 + kss];
                bb.u[1] = *(const uint4*)&Vt[col * 64 + kss + 16];
                o[nt] = wmma_bf16(a, bb, o[nt]);
            }
        }
    }

    // finalize: O / l, bf16 store into [B*T, 1024]
#pragma unroll
    for (int v = 0; v < 8; v++) {
        float inv = 1.f / l[v];
        int qrow = q0 + wid * 16 + v + 8 * half;
#pragma unroll
        for (int nt = 0; nt < 4; nt++)
            O[(rowbase + qrow) * C + colh + nt * 16 + l16] = f2bf(o[nt][v] * inv);
    }
}

// ---------------------------------------------------------------------------
extern "C" void kernel_launch(void* const* d_in, const int* in_sizes, int n_in,
                              void* d_out, int out_size, void* d_ws, size_t ws_size,
                              hipStream_t stream) {
    const float* x   = (const float*)d_in[0];
    const float* g1  = (const float*)d_in[1];
    const float* be1 = (const float*)d_in[2];
    const float* Wq  = (const float*)d_in[3];
    const float* bq  = (const float*)d_in[4];
    const float* Wk  = (const float*)d_in[5];
    const float* bk  = (const float*)d_in[6];
    const float* Wv  = (const float*)d_in[7];
    const float* bv  = (const float*)d_in[8];
    const float* Wo  = (const float*)d_in[9];
    const float* bo  = (const float*)d_in[10];
    const float* g2  = (const float*)d_in[11];
    const float* be2 = (const float*)d_in[12];
    const float* W1  = (const float*)d_in[13];
    const float* b1  = (const float*)d_in[14];
    const float* W2  = (const float*)d_in[15];
    const float* b2  = (const float*)d_in[16];

    const int B = 2, T = 2048, C = 1024, FF = 4096, M = B * T;

    char* ws = (char*)d_ws;
    size_t off = 0;
    auto carve = [&](size_t bytes) -> void* {
        void* p = ws + off;
        off += (bytes + 255) & ~(size_t)255;
        return p;
    };
    u16*   ln1o  = (u16*)carve((size_t)M * C * 2);
    u16*   Qb    = (u16*)carve((size_t)M * C * 2);
    u16*   Kb    = (u16*)carve((size_t)M * C * 2);
    u16*   Vb    = (u16*)carve((size_t)M * C * 2);
    u16*   attnb = (u16*)carve((size_t)M * C * 2);
    float* x1    = (float*)carve((size_t)M * C * 4);
    u16*   ln2o  = (u16*)carve((size_t)M * C * 2);
    u16*   h1    = (u16*)carve((size_t)M * FF * 2);
    (void)ws_size; (void)in_sizes; (void)n_in; (void)out_size;

    // 1. LN1
    layernorm_bf16<<<M, 256, 0, stream>>>(x, g1, be1, ln1o);
    // 2. Q/K/V projections
    dim3 gQKV(C / 128, M / 128);
    gemm_bf16<0><<<gQKV, 256, 0, stream>>>(ln1o, Wq, bq, nullptr, Qb, M, C, C);
    gemm_bf16<0><<<gQKV, 256, 0, stream>>>(ln1o, Wk, bk, nullptr, Kb, M, C, C);
    gemm_bf16<0><<<gQKV, 256, 0, stream>>>(ln1o, Wv, bv, nullptr, Vb, M, C, C);
    // 3. causal flash attention
    flash_attn<<<dim3(T / 64, 16, B), 128, 0, stream>>>(Qb, Kb, Vb, attnb, T);
    // 4. out-proj + residual
    gemm_bf16<1><<<gQKV, 256, 0, stream>>>(attnb, Wo, bo, x, x1, M, C, C);
    // 5. LN2
    layernorm_bf16<<<M, 256, 0, stream>>>(x1, g2, be2, ln2o);
    // 6. FF1 + GELU
    gemm_bf16<2><<<dim3(FF / 128, M / 128), 256, 0, stream>>>(ln2o, W1, b1, nullptr, h1, M, FF, C);
    // 7. FF2 + residual -> final fp32 output
    gemm_bf16<1><<<dim3(C / 128, M / 128), 256, 0, stream>>>(h1, W2, b2, x1, (float*)d_out, M, C, FF);
}
